// GAT_36249523978473
// MI455X (gfx1250) — matrix operand order repaired and checked by reading, not compiled, not run on hardware
//
#include <hip/hip_runtime.h>

typedef __bf16 bf16_t;
typedef __attribute__((ext_vector_type(16))) __bf16 v16bf;
typedef __attribute__((ext_vector_type(8)))  __bf16 v8bf;
typedef __attribute__((ext_vector_type(8)))  float  v8f;

#define GAT_ALPHA 0.2f
#define CSTRIDE 40   // column stride (elements) for column-major LDS tiles

#if defined(__has_builtin)
#if __has_builtin(__builtin_amdgcn_tensor_load_to_lds) && __has_builtin(__builtin_amdgcn_s_wait_tensorcnt)
#define USE_TDM 1
#endif
#endif

__device__ __forceinline__ float lrelu(float v) { return v > 0.f ? v : GAT_ALPHA * v; }
__device__ __forceinline__ float eluf (float v) { return v > 0.f ? v : (__expf(v) - 1.f); }

#ifdef USE_TDM
typedef __attribute__((ext_vector_type(4))) unsigned int u32x4;
typedef __attribute__((ext_vector_type(8))) int i32x8;
typedef __attribute__((ext_vector_type(4))) int i32x4;

// Issue a 2D TENSOR_LOAD_TO_LDS: tile = rows x rowElems, row stride = strideElems.
// dsizeCode: 0=1B,1=2B,2=4B,3=8B.  D# packing per CDNA5 ISA ch.8.
// clang-23 form: (u32x4 g0, i32x8 g1, i32x4 g2, i32x4 g3, i32x8, i32 cpol)
__device__ __forceinline__ void tdm_load_2d(void* lds, const void* gaddr,
                                            unsigned rowElems, unsigned rows,
                                            unsigned strideElems, unsigned dsizeCode)
{
  unsigned long long ga = (unsigned long long)(size_t)gaddr;
  unsigned lo = (unsigned)(size_t)lds;                 // flat LDS addr: low 32 = LDS offset
  u32x4 g0;
  g0[0] = 1u;                                          // count=1, user mode
  g0[1] = lo;                                          // lds_addr
  g0[2] = (unsigned)ga;                                // global_addr[31:0]
  g0[3] = (unsigned)((ga >> 32) & 0x1FFFFFFull) | (2u << 30); // addr[56:32] | type=2
  i32x8 g1;
  g1[0] = (int)(dsizeCode << 16);                      // workgroup_mask=0 | data_size
  g1[1] = (int)((rowElems & 0xFFFFu) << 16);           // tensor_dim0[15:0] @ bits 63:48
  g1[2] = (int)((rowElems >> 16) | ((rows & 0xFFFFu) << 16)); // dim0 hi | tensor_dim1 lo
  g1[3] = (int)((rows >> 16) | ((rowElems & 0xFFFFu) << 16)); // dim1 hi | tile_dim0
  g1[4] = (int)(rows & 0xFFFFu);                       // tile_dim1 | tile_dim2=0 (2D)
  g1[5] = (int)strideElems;                            // tensor_dim0_stride[31:0]
  g1[6] = 0;                                           // stride hi | dim1_stride lo
  g1[7] = 0;
  i32x4 z4 = {0, 0, 0, 0};
  i32x8 z8 = {0, 0, 0, 0, 0, 0, 0, 0};
  __builtin_amdgcn_tensor_load_to_lds(g0, g1, z4, z4, z8, 0);
}
#endif

__device__ __forceinline__ v16bf frag16(const bf16_t* p) {
  v8bf lo = *(const v8bf*)(p);
  v8bf hi = *(const v8bf*)(p + 16);
  return __builtin_shufflevector(lo, hi, 0,1,2,3,4,5,6,7,8,9,10,11,12,13,14,15);
}

// ---------------------------------------------------------------------------
// Cb[h][M,DC] (bf16) = A[M,K] (f32) @ B[h][K,DC] (f32), f32 accum via WMMA.
// 1 wave per 16-row tile per head.  DC in {16,64,128}, K multiple of 32.
// ---------------------------------------------------------------------------
template<int DC>
__global__ __launch_bounds__(32) void k_gemm_bf16(
    const float* __restrict__ A, const float* __restrict__ B, bf16_t* __restrict__ Cb,
    int M, int K)
{
  constexpr int NT = DC / 16;
  __shared__ __align__(64) bf16_t sA[16 * CSTRIDE];   // row-major 16x32 (+pad)
  __shared__ __align__(64) bf16_t sBc[DC * CSTRIDE];  // column-major 32xDC
#ifdef USE_TDM
  __shared__ __align__(16) float sRawF[32 * DC];      // TDM landing pad (f32 tile)
#endif
  const int lane = threadIdx.x;
  const int i0   = blockIdx.x * 16;
  const int h    = blockIdx.y;
  const float* Bh = B + (size_t)h * K * DC;
  bf16_t*      Ch = Cb + (size_t)h * M * DC;

  v8f acc[NT];
  #pragma unroll
  for (int t = 0; t < NT; ++t) acc[t] = (v8f){0.f,0.f,0.f,0.f,0.f,0.f,0.f,0.f};

  const int kbase = (lane >= 16) ? 8 : 0;   // A fragment K-group select
  const int mrow  = lane & 15;              // A fragment row
  const int brow0 = (lane < 16) ? 0 : 8;    // B fragment K-group select
  const int bn    = lane & 15;              // B fragment column

  for (int k0 = 0; k0 < K; k0 += 32) {
#ifdef USE_TDM
    // DMA the 32xDC f32 weight tile straight into LDS (no VGPR round-trip).
    tdm_load_2d(sRawF, Bh + (size_t)k0 * DC, DC, 32, DC, 2);
#endif
    // stage A tile 16x32 (coalesced f32 loads -> bf16 LDS, row-major)
    #pragma unroll
    for (int e = 0; e < 16; ++e) {
      int idx = e * 32 + lane;
      int r = idx >> 5, c = idx & 31;
      sA[r * CSTRIDE + c] = (bf16_t)A[(size_t)(i0 + r) * K + k0 + c];
    }
#ifdef USE_TDM
    __builtin_amdgcn_s_wait_tensorcnt(0);
    __syncthreads();
    // convert f32 -> bf16, transpose into column-major fragment layout
    #pragma unroll
    for (int r = 0; r < 32; ++r)
      #pragma unroll
      for (int c = lane, it = 0; it < DC / 32 + (DC < 32 ? 1 : 0); c += 32, ++it)
        if (c < DC) sBc[c * CSTRIDE + r] = (bf16_t)sRawF[r * DC + c];
#else
    for (int r = 0; r < 32; ++r)
      for (int c = lane; c < DC; c += 32)
        sBc[c * CSTRIDE + r] = (bf16_t)Bh[(size_t)(k0 + r) * DC + c];
#endif
    __syncthreads();

    // A fragment: element e -> K = e + (e>=8?8:0) + kbase  (two contiguous runs of 8)
    v8bf alo = *(const v8bf*)(&sA[mrow * CSTRIDE + kbase]);
    v8bf ahi = *(const v8bf*)(&sA[mrow * CSTRIDE + kbase + 16]);
    v16bf a = __builtin_shufflevector(alo, ahi, 0,1,2,3,4,5,6,7,8,9,10,11,12,13,14,15);
    #pragma unroll
    for (int t = 0; t < NT; ++t) {
      // B fragment: element e -> K = brow0 + e + (e>=8?8:0), N = 16t + bn  (two runs of 8 in col-major)
      v16bf b = frag16(&sBc[(16 * t + bn) * CSTRIDE + brow0]);
      acc[t] = __builtin_amdgcn_wmma_f32_16x16x32_bf16(
          false, a, false, b, (short)0, acc[t], false, false);
    }
    __syncthreads();
  }

  // C/D layout: vgpr r -> M = r + (lane>=16?8:0), N = lane&15
  const int mr = (lane >= 16) ? 8 : 0;
  #pragma unroll
  for (int t = 0; t < NT; ++t)
    #pragma unroll
    for (int r = 0; r < 8; ++r)
      Ch[(size_t)(i0 + r + mr) * DC + 16 * t + bn] = (bf16_t)acc[t][r];
}

// ---------------------------------------------------------------------------
// f1[h][i] = h_i . a1,  f2[h][i] = h_i . a2
// ---------------------------------------------------------------------------
__global__ void k_scores(const bf16_t* __restrict__ Hb, const float* __restrict__ a,
                         float* __restrict__ f1, float* __restrict__ f2,
                         int Nn, int Dc, int NH)
{
  int t = blockIdx.x * blockDim.x + threadIdx.x;
  if (t >= NH * Nn) return;
  int h = t / Nn, i = t - h * Nn;
  const bf16_t* hrow = Hb + ((size_t)h * Nn + i) * Dc;
  const float* a1 = a + (size_t)h * 2 * Dc;
  const float* a2 = a1 + Dc;
  float s1 = 0.f, s2 = 0.f;
  for (int d = 0; d < Dc; ++d) {
    float v = (float)hrow[d];
    s1 += v * a1[d];
    s2 += v * a2[d];
  }
  f1[(size_t)h * Nn + i] = s1;
  f2[(size_t)h * Nn + i] = s2;
}

// ---------------------------------------------------------------------------
// Online masked softmax stats: per row i, all NH heads in ONE adj pass.
// ---------------------------------------------------------------------------
template<int NH>
__global__ __launch_bounds__(32) void k_stats(
    const int* __restrict__ adj, const float* __restrict__ f1, const float* __restrict__ f2,
    float* __restrict__ mo, float* __restrict__ so, int Nn)
{
  const int i = blockIdx.x;
  const int lane = threadIdx.x;
  float f1h[NH], ml[NH], sl[NH];
  #pragma unroll
  for (int h = 0; h < NH; ++h) { f1h[h] = f1[(size_t)h * Nn + i]; ml[h] = -3e38f; sl[h] = 0.f; }
  const int* arow = adj + (size_t)i * Nn;
  for (int j = lane; j < Nn; j += 32) {
    __builtin_prefetch(arow + j + 512, 0, 0);
    int av = arow[j];
    if (av > 0) {
      #pragma unroll
      for (int h = 0; h < NH; ++h) {
        float e = lrelu(f1h[h] + f2[(size_t)h * Nn + j]);
        if (e > ml[h]) { sl[h] = sl[h] * __expf(ml[h] - e) + 1.f; ml[h] = e; }
        else            sl[h] += __expf(e - ml[h]);
      }
    }
  }
  #pragma unroll
  for (int h = 0; h < NH; ++h) {
    float m = ml[h], s = sl[h];
    #pragma unroll
    for (int off = 16; off > 0; off >>= 1) {
      float m2 = __shfl_xor(m, off, 32);
      float s2 = __shfl_xor(s, off, 32);
      float M2 = fmaxf(m, m2);
      s = s * __expf(m - M2) + s2 * __expf(m2 - M2);
      m = M2;
    }
    if (lane == 0) { mo[(size_t)h * Nn + i] = m; so[(size_t)h * Nn + i] = s; }
  }
}

// ---------------------------------------------------------------------------
// Flash-style masked attention aggregation with WMMA.
//   MODE 0: elu(val + resid) into concat slot  (layer 1, identity residual)
//   MODE 1: elu(val) into concat slot          (layer 2)
//   MODE 2: raw val                            (layer 3)
// ---------------------------------------------------------------------------
template<int DC, int MODE>
__global__ __launch_bounds__(32) void k_agg(
    const bf16_t* __restrict__ Hb, const float* __restrict__ f1, const float* __restrict__ f2,
    const float* __restrict__ mrowv, const float* __restrict__ srowv,
    const int* __restrict__ adj, const float* __restrict__ resid,
    float* __restrict__ outp, int Nn, int Hn)
{
  constexpr int NT = DC / 16;
  __shared__ __align__(64) bf16_t sHc[DC * CSTRIDE];  // column-major 32xDC block
#ifdef USE_TDM
  __shared__ __align__(16) bf16_t sRawH[32 * DC];     // TDM landing pad (bf16 tile)
#endif
  const int lane = threadIdx.x;
  const int i0 = blockIdx.x * 16;
  const int h  = blockIdx.y;
  const bf16_t* Hbh = Hb + (size_t)h * Nn * DC;
  const int irow = i0 + (lane & 15);
  const float f1i = f1[(size_t)h * Nn + irow];
  const float mi  = mrowv[(size_t)h * Nn + irow];
  const float* f2h = f2 + (size_t)h * Nn;
  const int* arow = adj + (size_t)irow * Nn;
  const int kbase = (lane >= 16) ? 8 : 0;
  const int brow0 = (lane < 16) ? 0 : 8;
  const int bn  = lane & 15;

  v8f acc[NT];
  #pragma unroll
  for (int t = 0; t < NT; ++t) acc[t] = (v8f){0.f,0.f,0.f,0.f,0.f,0.f,0.f,0.f};

  for (int j0 = 0; j0 < Nn; j0 += 32) {
#ifdef USE_TDM
    // DMA the 32xDC bf16 feature block into LDS.
    tdm_load_2d(sRawH, Hbh + (size_t)j0 * DC, DC, 32, DC, 1);
#endif
    __builtin_prefetch(arow + j0 + 128, 0, 0);

    // P tile in A-fragment layout: e -> j = j0 + kbase + e (e<8), +16 for e>=8
    const int4   q0 = *(const int4*)  (arow + j0 + kbase);
    const int4   q1 = *(const int4*)  (arow + j0 + kbase + 4);
    const int4   q2 = *(const int4*)  (arow + j0 + 16 + kbase);
    const int4   q3 = *(const int4*)  (arow + j0 + 16 + kbase + 4);
    const float4 g0 = *(const float4*)(f2h + j0 + kbase);
    const float4 g1 = *(const float4*)(f2h + j0 + kbase + 4);
    const float4 g2 = *(const float4*)(f2h + j0 + 16 + kbase);
    const float4 g3 = *(const float4*)(f2h + j0 + 16 + kbase + 4);
    const int   am[16] = {q0.x,q0.y,q0.z,q0.w, q1.x,q1.y,q1.z,q1.w,
                          q2.x,q2.y,q2.z,q2.w, q3.x,q3.y,q3.z,q3.w};
    const float gm[16] = {g0.x,g0.y,g0.z,g0.w, g1.x,g1.y,g1.z,g1.w,
                          g2.x,g2.y,g2.z,g2.w, g3.x,g3.y,g3.z,g3.w};
    v16bf p;
    #pragma unroll
    for (int e = 0; e < 16; ++e) {
      float pe = (am[e] > 0) ? __expf(lrelu(f1i + gm[e]) - mi) : 0.f;
      p[e] = (bf16_t)pe;
    }

#ifdef USE_TDM
    __builtin_amdgcn_s_wait_tensorcnt(0);
    __syncthreads();
    // transpose LDS->LDS into column-major fragment layout
    #pragma unroll
    for (int r = 0; r < 32; ++r)
      #pragma unroll
      for (int c = lane, it = 0; it < DC / 32 + (DC < 32 ? 1 : 0); c += 32, ++it)
        if (c < DC) sHc[c * CSTRIDE + r] = sRawH[r * DC + c];
#else
    for (int r = 0; r < 32; ++r)
      for (int c = lane; c < DC; c += 32)
        sHc[c * CSTRIDE + r] = Hbh[(size_t)(j0 + r) * DC + c];
#endif
    __syncthreads();

    #pragma unroll
    for (int t = 0; t < NT; ++t) {
      v16bf b = frag16(&sHc[(16 * t + bn) * CSTRIDE + brow0]);
      acc[t] = __builtin_amdgcn_wmma_f32_16x16x32_bf16(
          false, p, false, b, (short)0, acc[t], false, false);
    }
    __syncthreads();
  }

  const int mr = (lane >= 16) ? 8 : 0;
  const int ldo = Hn * DC;
  #pragma unroll
  for (int r = 0; r < 8; ++r) {
    const int row = i0 + r + mr;
    float sv_ = srowv[(size_t)h * Nn + row];
    float inv = sv_ > 0.f ? 1.f / sv_ : 0.f;
    #pragma unroll
    for (int t = 0; t < NT; ++t) {
      float val = acc[t][r] * inv;
      int col = 16 * t + bn;
      if (MODE == 0)      val = eluf(val + resid[(size_t)row * DC + col]);
      else if (MODE == 1) val = eluf(val);
      outp[(size_t)row * ldo + h * DC + col] = val;
    }
  }
}

// ---------------------------------------------------------------------------
// out = log_softmax(elu(h3), axis=1), C = 16
// ---------------------------------------------------------------------------
__global__ void k_logsoftmax(const float* __restrict__ h3, float* __restrict__ outp, int Nn)
{
  int i = blockIdx.x * blockDim.x + threadIdx.x;
  if (i >= Nn) return;
  float v[16];
  float mx = -3e38f;
  #pragma unroll
  for (int k = 0; k < 16; ++k) { v[k] = eluf(h3[(size_t)i * 16 + k]); mx = fmaxf(mx, v[k]); }
  float s = 0.f;
  #pragma unroll
  for (int k = 0; k < 16; ++k) s += __expf(v[k] - mx);
  float l = __logf(s) + mx;
  #pragma unroll
  for (int k = 0; k < 16; ++k) outp[(size_t)i * 16 + k] = v[k] - l;
}

// ---------------------------------------------------------------------------
extern "C" void kernel_launch(void* const* d_in, const int* in_sizes, int n_in,
                              void* d_out, int out_size, void* d_ws, size_t ws_size,
                              hipStream_t stream)
{
  (void)in_sizes; (void)n_in; (void)out_size; (void)ws_size;
  const int N = 4096, F = 128, HID = 64, OUT = 16, H = 4;
  const float* x     = (const float*)d_in[0];
  const int*   adj   = (const int*)  d_in[1];
  const float* W_in  = (const float*)d_in[2];
  const float* a_in  = (const float*)d_in[3];
  const float* W_hid = (const float*)d_in[4];
  const float* a_hid = (const float*)d_in[5];
  const float* W_out = (const float*)d_in[6];
  const float* a_out = (const float*)d_in[7];
  float* out = (float*)d_out;

  char* w = (char*)d_ws;
  size_t off = 0;
  auto take = [&](size_t bytes) -> void* {
    void* p = w + off;
    off += (bytes + 255) & ~(size_t)255;
    return p;
  };
  bf16_t* Hb1  = (bf16_t*)take((size_t)H * N * F * 2);
  float*  out1 = (float*) take((size_t)N * (H * F) * 4);
  bf16_t* Hb2  = (bf16_t*)take((size_t)H * N * HID * 2);
  float*  out2 = (float*) take((size_t)N * (H * HID) * 4);
  bf16_t* Hb3  = (bf16_t*)take((size_t)N * OUT * 2);
  float*  h3   = (float*) take((size_t)N * OUT * 4);
  float*  f1   = (float*) take((size_t)H * N * 4);
  float*  f2   = (float*) take((size_t)H * N * 4);
  float*  mv   = (float*) take((size_t)H * N * 4);
  float*  sv   = (float*) take((size_t)H * N * 4);

  dim3 b32(32), b256(256);

  // ---- Layer 1: input attention (concat, identity residual, ELU) ----
  k_gemm_bf16<128><<<dim3(N / 16, H), b32, 0, stream>>>(x, W_in, Hb1, N, F);
  k_scores<<<dim3((H * N + 255) / 256), b256, 0, stream>>>(Hb1, a_in, f1, f2, N, F, H);
  k_stats<4><<<dim3(N), b32, 0, stream>>>(adj, f1, f2, mv, sv, N);
  k_agg<128, 0><<<dim3(N / 16, H), b32, 0, stream>>>(Hb1, f1, f2, mv, sv, adj, x, out1, N, H);

  // ---- Layer 2: hidden attention (concat, ELU) ----
  k_gemm_bf16<64><<<dim3(N / 16, H), b32, 0, stream>>>(out1, W_hid, Hb2, N, H * F);
  k_scores<<<dim3((H * N + 255) / 256), b256, 0, stream>>>(Hb2, a_hid, f1, f2, N, HID, H);
  k_stats<4><<<dim3(N), b32, 0, stream>>>(adj, f1, f2, mv, sv, N);
  k_agg<64, 1><<<dim3(N / 16, H), b32, 0, stream>>>(Hb2, f1, f2, mv, sv, adj, nullptr, out2, N, H);

  // ---- Layer 3: output attention (single head, raw) ----
  k_gemm_bf16<16><<<dim3(N / 16, 1), b32, 0, stream>>>(out2, W_out, Hb3, N, H * HID);
  k_scores<<<dim3((N + 255) / 256), b256, 0, stream>>>(Hb3, a_out, f1, f2, N, OUT, 1);
  k_stats<1><<<dim3(N), b32, 0, stream>>>(adj, f1, f2, mv, sv, N);
  k_agg<16, 2><<<dim3(N / 16, 1), b32, 0, stream>>>(Hb3, f1, f2, mv, sv, adj, nullptr, h3, N, 1);

  // ---- ELU + log_softmax ----
  k_logsoftmax<<<dim3((N + 255) / 256), b256, 0, stream>>>(h3, out, N);
}
